// MoEBlock_32246614458988
// MI455X (gfx1250) — matrix operand order repaired and checked by reading, not compiled, not run on hardware
//
#include <hip/hip_runtime.h>
#include <hip/hip_bf16.h>
#include <math.h>

typedef __attribute__((ext_vector_type(16))) _Float16 v16h;
typedef __attribute__((ext_vector_type(8)))  _Float16 v8h;
typedef __attribute__((ext_vector_type(8)))  float    v8f;

#define D_DIM 1024
#define H_DIM 4096
#define E_NUM 8
#define TM    16

__device__ __forceinline__ v16h mk16(v8h lo, v8h hi) {
  return __builtin_shufflevector(lo, hi, 0,1,2,3,4,5,6,7,8,9,10,11,12,13,14,15);
}

// CDNA5 async copy: 16B global -> LDS, tracked by ASYNCcnt (no VGPR round-trip).
__device__ __forceinline__ void async_copy16(unsigned lds_off, int g_off,
                                             unsigned long long sbase) {
  asm volatile("global_load_async_to_lds_b128 %0, %1, %2 offset:0"
               :: "v"(lds_off), "v"(g_off), "s"(sbase) : "memory");
}
template <int N>
__device__ __forceinline__ void wait_async() {
  asm volatile("s_wait_asynccnt %0" :: "i"(N) : "memory");
}
__device__ __forceinline__ unsigned lds_addr(const void* p) {
  return (unsigned)(unsigned long long)p;   // low 32 bits of shared aperture = LDS offset
}

__global__ __launch_bounds__(256) void moe_init(int* counts) {
  if (threadIdx.x < E_NUM) counts[threadIdx.x] = 0;
}

// ---------------------------------------------------------------------------
// One-time repack: fp32 W1/W2 -> f16 tiles [e][k/32][n][32] (B-fragment order).
// W1: e(8) x kb(32) x nc(32) blocks; W2: e(8) x kb(128) x nc(8) blocks.
// ---------------------------------------------------------------------------
__global__ __launch_bounds__(256) void moe_convert_w(
    const float* __restrict__ W1, const float* __restrict__ W2,
    _Float16* __restrict__ W1h, _Float16* __restrict__ W2h) {
  __shared__ float tile[32][132];
  const int bid = blockIdx.x;
  const float* src;
  _Float16* dst;
  int N;
  if (bid < 8192) {
    int e = bid >> 10, rem = bid & 1023;
    int kb = rem >> 5, nc = rem & 31, n0 = nc * 128;
    src = W1 + ((size_t)e * D_DIM + kb * 32) * H_DIM + n0;
    dst = W1h + (((size_t)e * 32 + kb) * H_DIM + n0) * 32;
    N = H_DIM;
  } else {
    int b2 = bid - 8192;
    int e = b2 >> 10, rem = b2 & 1023;
    int kb = rem >> 3, nc = rem & 7, n0 = nc * 128;
    src = W2 + ((size_t)e * H_DIM + kb * 32) * D_DIM + n0;
    dst = W2h + (((size_t)e * 128 + kb) * D_DIM + n0) * 32;
    N = D_DIM;
  }
  const int tid = threadIdx.x;
  for (int idx = tid; idx < 32 * 128; idx += 256) {
    int kk = idx >> 7, n = idx & 127;
    tile[kk][n] = src[(size_t)kk * N + n];
  }
  __syncthreads();
  for (int idx = tid; idx < 512; idx += 256) {   // 128 n-rows x 4 x 16B
    int n = idx >> 2, part = idx & 3;
    v8h v;
#pragma unroll
    for (int j = 0; j < 8; ++j) v[j] = (_Float16)tile[part * 8 + j][n];
    *(v8h*)&dst[(size_t)n * 32 + part * 8] = v;
  }
}

// One block per token: router logits, top-2 + softmax, expert list append.
__global__ __launch_bounds__(256) void moe_router(
    const float* __restrict__ x, const float* __restrict__ Wr,
    const float* __restrict__ br, int* __restrict__ counts,
    float* __restrict__ weights, int* __restrict__ list,
    int4* __restrict__ tinfo, int T) {
  const int t = blockIdx.x;
  const int tid = threadIdx.x;
  float acc[E_NUM];
#pragma unroll
  for (int e = 0; e < E_NUM; ++e) acc[e] = 0.0f;
  const float* xr = x + (size_t)t * D_DIM;
  const int d0 = tid * 4;
#pragma unroll
  for (int i = 0; i < 4; ++i) {
    float xv = xr[d0 + i];
#pragma unroll
    for (int e = 0; e < E_NUM; ++e) acc[e] += xv * Wr[(d0 + i) * E_NUM + e];
  }
  __shared__ float red[256][E_NUM];
#pragma unroll
  for (int e = 0; e < E_NUM; ++e) red[tid][e] = acc[e];
  __syncthreads();
  for (int s = 128; s > 0; s >>= 1) {
    if (tid < s) {
#pragma unroll
      for (int e = 0; e < E_NUM; ++e) red[tid][e] += red[tid + s][e];
    }
    __syncthreads();
  }
  if (tid == 0) {
    float lg[E_NUM];
#pragma unroll
    for (int e = 0; e < E_NUM; ++e) lg[e] = red[0][e] + br[e];
    int i1 = 0;
#pragma unroll
    for (int e = 1; e < E_NUM; ++e) if (lg[e] > lg[i1]) i1 = e;
    int i2 = (i1 == 0) ? 1 : 0;
#pragma unroll
    for (int e = 0; e < E_NUM; ++e) if (e != i1 && lg[e] > lg[i2]) i2 = e;
    const float vmax = lg[i1];
    const float thresh = lg[i2];
    float w[E_NUM];
    float Z = 0.0f;
#pragma unroll
    for (int e = 0; e < E_NUM; ++e) {
      w[e] = (lg[e] >= thresh) ? expf(lg[e] - vmax) : 0.0f;
      Z += w[e];
    }
    const float invZ = 1.0f / Z;
#pragma unroll
    for (int e = 0; e < E_NUM; ++e) weights[t * E_NUM + e] = w[e] * invZ;
    int p0 = atomicAdd(&counts[i1], 1);
    int p1 = atomicAdd(&counts[i2], 1);
    list[i1 * T + p0] = t;
    list[i2 * T + p1] = t;
    tinfo[t] = make_int4(i1, p0, i2, p1);
  }
}

// One block per (expert, 16-token tile). Fused FF1 -> GELU -> FF2 with
// f16 WMMA (fp32 accumulate); writes expert rows (+b2) to the slot buffer.
// F16W=1: pre-packed f16 weights, double-buffered async staging (ASYNCcnt).
// F16W=0: fallback, stage fp32 weights through VGPRs with conversion.
template <int F16W>
__global__ __launch_bounds__(256) void moe_ffn(
    const float* __restrict__ x, const float* __restrict__ W1,
    const float* __restrict__ b1, const float* __restrict__ W2,
    const float* __restrict__ b2, const int* __restrict__ counts,
    const int* __restrict__ list, float* __restrict__ ebuf,
    const _Float16* __restrict__ W1h, const _Float16* __restrict__ W2h,
    int T) {
  const int TPE = T / TM;
  const int e    = blockIdx.x / TPE;
  const int tile = blockIdx.x % TPE;
  const int cnt  = counts[e];
  const int r0   = tile * TM;
  if (r0 >= cnt) return;
  int base = 0;
  for (int i = 0; i < e; ++i) base += counts[i];

  __shared__ __align__(16) _Float16 Xs[TM][1032];      // x tile, f16 (padded)
  __shared__ __align__(16) _Float16 W1s[2][128][40];   // W1 block, double-buffered
  __shared__ __align__(16) _Float16 W2s[1024][40];     // W2 block, [n][k]
  __shared__ __align__(16) _Float16 Hs[TM][136];       // gelu(h) chunk
  __shared__ int toks[TM];

  const int tid = threadIdx.x;
  if (tid < TM) toks[tid] = (r0 + tid < cnt) ? list[e * T + r0 + tid] : -1;
  __syncthreads();
  for (int idx = tid; idx < TM * D_DIM; idx += 256) {
    int r = idx >> 10, d = idx & (D_DIM - 1);
    int tk = toks[r];
    Xs[r][d] = (tk >= 0) ? (_Float16)x[(size_t)tk * D_DIM + d] : (_Float16)0.0f;
  }
  __syncthreads();

  const int wv   = tid >> 5;
  const int lane = tid & 31;
  const int n16  = lane & 15;
  const int hi   = lane >> 4;
  const int lo   = hi * 8;

  const float* W1e = W1 + (size_t)e * D_DIM * H_DIM;
  const float* W2e = W2 + (size_t)e * H_DIM * D_DIM;
  const unsigned long long w1h_base = (unsigned long long)W1h;
  const unsigned long long w2h_base = (unsigned long long)W2h;
  unsigned l1b[2] = {lds_addr(&W1s[0][0][0]), lds_addr(&W1s[1][0][0])};
  const unsigned l2 = lds_addr(&W2s[0][0]);
  // per-thread fixed (n,part) for 16B async chunks
  const int an = tid >> 2, apart = tid & 3;
  const unsigned aoff = (unsigned)(an * 80 + apart * 16);

  const v8f zero8 = {0.f, 0.f, 0.f, 0.f, 0.f, 0.f, 0.f, 0.f};
  v8f oacc[8];
#pragma unroll
  for (int nb = 0; nb < 8; ++nb) oacc[nb] = zero8;

  for (int hc = 0; hc < H_DIM / 128; ++hc) {
    const int hbase = hc * 128;
    // -------- FF1: this wave computes h columns [wv*16, wv*16+16)
    v8f hacc = zero8;
    if constexpr (F16W) {   // prime the pipeline: stage k-step 0 into buf 0
#pragma unroll
      for (int r = 0; r < 2; ++r) {
        int n = an + r * 64;
        int goff = (((e * 32 + 0) * H_DIM + hbase + n) * 32 + apart * 8) * 2;
        async_copy16(l1b[0] + aoff + (unsigned)(r * 64 * 80), goff, w1h_base);
      }
    }
    for (int ks = 0; ks < D_DIM / 32; ++ks) {
      int cur = 0;
      if constexpr (F16W) {
        cur = ks & 1;
        if (ks + 1 < D_DIM / 32) {
          // issue next step's group into the other buffer, then wait for
          // the current step's group only (2 in flight from next step).
#pragma unroll
          for (int r = 0; r < 2; ++r) {
            int n = an + r * 64;
            int goff = (((e * 32 + ks + 1) * H_DIM + hbase + n) * 32 + apart * 8) * 2;
            async_copy16(l1b[1 - cur] + aoff + (unsigned)(r * 64 * 80), goff, w1h_base);
          }
          wait_async<2>();
        } else {
          wait_async<0>();
        }
      } else {
        const int kb = ks * 32;
        for (int idx = tid; idx < 32 * 128; idx += 256) {
          int n = idx & 127, kk = idx >> 7;
          W1s[0][n][kk] = (_Float16)W1e[(size_t)(kb + kk) * H_DIM + hbase + n];
        }
      }
      __syncthreads();
      const int kb = ks * 32;
      v8h a0 = *(const v8h*)&Xs[n16][kb + lo];
      v8h a1 = *(const v8h*)&Xs[n16][kb + 16 + lo];
      v8h bb0 = *(const v8h*)&W1s[cur][wv * 16 + n16][lo];
      v8h bb1 = *(const v8h*)&W1s[cur][wv * 16 + n16][16 + lo];
      hacc = __builtin_amdgcn_wmma_f32_16x16x32_f16(
          false, mk16(a0, a1), false, mk16(bb0, bb1), (short)0, hacc, false, false);
      __syncthreads();
    }
    // bias + exact GELU, store chunk to LDS
    {
      float bv = b1[(size_t)e * H_DIM + hbase + wv * 16 + n16];
#pragma unroll
      for (int v = 0; v < 8; ++v) {
        float tv = hacc[v] + bv;
        float g = 0.5f * tv * (1.0f + erff(tv * 0.70710678118654752f));
        Hs[v + 8 * hi][wv * 16 + n16] = (_Float16)g;
      }
    }
    __syncthreads();
    // -------- FF2: oacc += Hs @ W2[hbase:hbase+128, :]; wave owns cols [wv*128, +128)
    for (int ks = 0; ks < 4; ++ks) {
      const int kb = ks * 32;
      if constexpr (F16W) {
        const int kb2 = hc * 4 + ks;   // global 32-row block index into W2h
#pragma unroll
        for (int r = 0; r < 16; ++r) {
          int n = an + r * 64;
          unsigned lds = l2 + aoff + (unsigned)(r * 64 * 80);
          int goff = (((e * 128 + kb2) * D_DIM + n) * 32 + apart * 8) * 2;
          async_copy16(lds, goff, w2h_base);
        }
        wait_async<0>();
      } else {
        for (int idx = tid; idx < 32 * 1024; idx += 256) {
          int n = idx & 1023, kk = idx >> 10;
          W2s[n][kk] = (_Float16)W2e[(size_t)(hbase + kb + kk) * D_DIM + n];
        }
      }
      __syncthreads();
      v8h a0 = *(const v8h*)&Hs[n16][kb + lo];
      v8h a1 = *(const v8h*)&Hs[n16][kb + 16 + lo];
      v16h a = mk16(a0, a1);
#pragma unroll
      for (int nb = 0; nb < 8; ++nb) {
        int col = wv * 128 + nb * 16 + n16;
        v8h bb0 = *(const v8h*)&W2s[col][lo];
        v8h bb1 = *(const v8h*)&W2s[col][16 + lo];
        oacc[nb] = __builtin_amdgcn_wmma_f32_16x16x32_f16(
            false, a, false, mk16(bb0, bb1), (short)0, oacc[nb], false, false);
      }
      __syncthreads();
    }
  }
  // epilogue: +b2, scatter valid rows into compact slot buffer
#pragma unroll
  for (int nb = 0; nb < 8; ++nb) {
    int col = wv * 128 + nb * 16 + n16;
    float b2v = b2[(size_t)e * D_DIM + col];
#pragma unroll
    for (int v = 0; v < 8; ++v) {
      int m = v + 8 * hi;
      if (r0 + m < cnt)
        ebuf[(size_t)(base + r0 + m) * D_DIM + col] = oacc[nb][v] + b2v;
    }
  }
}

// out[t] = w0*slot(e0) + w1*slot(e1)  (fixed order -> deterministic)
__global__ __launch_bounds__(256) void moe_combine(
    const float* __restrict__ weights, const int4* __restrict__ tinfo,
    const int* __restrict__ counts, const float* __restrict__ ebuf,
    float* __restrict__ out, int T) {
  const int t = blockIdx.x;
  const int4 info = tinfo[t];
  int base0 = 0, base1 = 0;
#pragma unroll
  for (int i = 0; i < E_NUM; ++i) {
    int c = counts[i];
    if (i < info.x) base0 += c;
    if (i < info.z) base1 += c;
  }
  const size_t s0 = (size_t)(base0 + info.y) * D_DIM;
  const size_t s1 = (size_t)(base1 + info.w) * D_DIM;
  const float w0 = weights[t * E_NUM + info.x];
  const float w1 = weights[t * E_NUM + info.z];
  for (int d = threadIdx.x; d < D_DIM; d += 256) {
    out[(size_t)t * D_DIM + d] = w0 * ebuf[s0 + d] + w1 * ebuf[s1 + d];
  }
}

extern "C" void kernel_launch(void* const* d_in, const int* in_sizes, int n_in,
                              void* d_out, int out_size, void* d_ws, size_t ws_size,
                              hipStream_t stream) {
  const float* x  = (const float*)d_in[0];
  const float* W1 = (const float*)d_in[1];
  const float* b1 = (const float*)d_in[2];
  const float* W2 = (const float*)d_in[3];
  const float* b2 = (const float*)d_in[4];
  const float* Wr = (const float*)d_in[5];
  const float* br = (const float*)d_in[6];
  float* out = (float*)d_out;
  const int T = in_sizes[0] / D_DIM;   // 4096 tokens

  // workspace layout (bytes):
  //   [0,32)          counts (int[8]), padded to 256
  //   [256, +32T)     weights  (T x 8 f32)
  //   [.., +32T)      list     (8 x T int)
  //   [.., +16T)      tinfo    (T x int4)
  //   [.., +8192T)    ebuf     (2T x 1024 f32 slot buffer)
  //   [.., +64MB)     W1h      (f16 repack, optional)
  //   [.., +64MB)     W2h      (f16 repack, optional)
  char* ws = (char*)d_ws;
  int*   counts  = (int*)ws;
  float* weights = (float*)(ws + 256);
  int*   list    = (int*)(ws + 256 + (size_t)T * E_NUM * 4);
  int4*  tinfo   = (int4*)(ws + 256 + 2 * (size_t)T * E_NUM * 4);
  float* ebuf    = (float*)(ws + 256 + 2 * (size_t)T * E_NUM * 4 + (size_t)T * 16);
  const size_t wh_bytes = (size_t)E_NUM * D_DIM * H_DIM * 2;   // 64MB each
  _Float16* W1h = (_Float16*)(ws + 256 + (size_t)T * 80 + (size_t)T * 8192);
  _Float16* W2h = W1h + (size_t)E_NUM * D_DIM * H_DIM;
  const size_t need = 256 + (size_t)T * 80 + (size_t)T * 8192 + 2 * wh_bytes;
  const int f16w = (ws_size >= need) ? 1 : 0;

  hipLaunchKernelGGL(moe_init, dim3(1), dim3(256), 0, stream, counts);
  if (f16w)
    hipLaunchKernelGGL(moe_convert_w, dim3(16384), dim3(256), 0, stream,
                       W1, W2, W1h, W2h);
  hipLaunchKernelGGL(moe_router, dim3(T), dim3(256), 0, stream,
                     x, Wr, br, counts, weights, list, tinfo, T);
  if (f16w)
    hipLaunchKernelGGL((moe_ffn<1>), dim3(E_NUM * (T / TM)), dim3(256), 0, stream,
                       x, W1, b1, W2, b2, counts, list, ebuf, W1h, W2h, T);
  else
    hipLaunchKernelGGL((moe_ffn<0>), dim3(E_NUM * (T / TM)), dim3(256), 0, stream,
                       x, W1, b1, W2, b2, counts, list, ebuf, W1h, W2h, T);
  hipLaunchKernelGGL(moe_combine, dim3(T), dim3(256), 0, stream,
                     weights, tinfo, counts, ebuf, out, T);
}
